// ConvLayer_30520037605802
// MI455X (gfx1250) — matrix operand order repaired and checked by reading, not compiled, not run on hardware
//
#include <hip/hip_runtime.h>
#include <hip/hip_bf16.h>
#include <math.h>

typedef __attribute__((ext_vector_type(16))) _Float16 v16h;
typedef __attribute__((ext_vector_type(8)))  float    v8f;

#define NATOMS 60000
#define MNBR   12
#define DF     64
#define NBRF   41
#define BCRY   1200
#define NACRY  50
#define C2     128
#define NMROWS (NATOMS * MNBR)   // 720000
#define EPSV   1e-5f

// workspace layout (float offsets)
#define WS_ATOMTF 0            // 60000*64      = 3,840,000 f
#define WS_PA     3840000      // 60000*128     = 7,680,000 f
#define WS_PB     11520000     // 60000*128     = 7,680,000 f
#define WS_NS     19200000     // 60000*64      = 3,840,000 f
#define WS_STATS  23040000     // 768 f: sum1[128] sq1[128] mu1[128] rstd1[128] sum2[64] sq2[64] mu2[64] rstd2[64]
#define WS_GATED  23040768     // _Float16[720000*128]

// ---------------------------------------------------------------------------
// K0: zero the stat accumulators (must be re-zeroed every call)
__global__ void k_zero_stats(float* __restrict__ stats) {
    int t = threadIdx.x;
    if (t < 768) stats[t] = 0.f;
}

// ---------------------------------------------------------------------------
// layernorm of one 64-row (biased variance, matches torch LayerNorm)
__device__ inline void ln_row(const float* in, float* out,
                              const float* __restrict__ g, const float* __restrict__ b) {
    float mu = 0.f;
    #pragma unroll 8
    for (int k = 0; k < DF; ++k) mu += in[k];
    mu *= (1.f / DF);
    float var = 0.f;
    #pragma unroll 8
    for (int k = 0; k < DF; ++k) { float d = in[k] - mu; var += d * d; }
    var *= (1.f / DF);
    float rs = rsqrtf(var + EPSV);
    #pragma unroll 8
    for (int k = 0; k < DF; ++k) out[k] = (in[k] - mu) * rs * g[k] + b[k];
}

// ---------------------------------------------------------------------------
// K1: per-crystal graph-transformer layer -> atom_tf [60000,64]
__global__ void __launch_bounds__(256) k_transformer(
    const float* __restrict__ atom_in, const float* __restrict__ adj,
    const float* __restrict__ w1, const float* __restrict__ b1,
    const float* __restrict__ w2, const float* __restrict__ b2,
    const float* __restrict__ g1, const float* __restrict__ be1,
    const float* __restrict__ g2, const float* __restrict__ be2,
    const int*   __restrict__ cidx, float* __restrict__ atom_tf)
{
    __shared__ float s_src[NACRY * DF];
    __shared__ float s_tmp[NACRY * DF];
    __shared__ float s_h1 [NACRY * DF];
    __shared__ float s_sc [NACRY * 52];
    __shared__ int   s_idx[NACRY];
    const int b = blockIdx.x, tid = threadIdx.x;

    for (int i = tid; i < NACRY; i += 256) s_idx[i] = cidx[b * NACRY + i];
    __syncthreads();
    for (int i = tid; i < NACRY * DF; i += 256)
        s_src[i] = atom_in[s_idx[i >> 6] * DF + (i & 63)];
    __syncthreads();

    // scores = (src @ src^T) * (1/sqrt(D)) * adj
    for (int p = tid; p < NACRY * NACRY; p += 256) {
        int i = p / NACRY, j = p - i * NACRY;
        float acc = 0.f;
        #pragma unroll 8
        for (int k = 0; k < DF; ++k) acc += s_src[i * DF + k] * s_src[j * DF + k];
        s_sc[i * 52 + j] = acc * 0.125f * adj[(b * NACRY + i) * NACRY + j];
    }
    __syncthreads();

    // softmax over rows (50 rows of 50)
    if (tid < NACRY) {
        float mx = -3.4e38f;
        for (int j = 0; j < NACRY; ++j) mx = fmaxf(mx, s_sc[tid * 52 + j]);
        float s = 0.f;
        for (int j = 0; j < NACRY; ++j) {
            float e = __expf(s_sc[tid * 52 + j] - mx);
            s_sc[tid * 52 + j] = e; s += e;
        }
        float inv = 1.f / s;
        for (int j = 0; j < NACRY; ++j) s_sc[tid * 52 + j] *= inv;
    }
    __syncthreads();

    // src2 = attn @ src ; residual
    for (int p = tid; p < NACRY * DF; p += 256) {
        int i = p >> 6, d = p & 63;
        float acc = 0.f;
        for (int j = 0; j < NACRY; ++j) acc += s_sc[i * 52 + j] * s_src[j * DF + d];
        s_tmp[p] = acc + s_src[p];
    }
    __syncthreads();
    if (tid < NACRY) ln_row(&s_tmp[tid * DF], &s_src[tid * DF], g1, be1);
    __syncthreads();

    // FFN: h1 = relu(src @ w1 + b1)
    for (int p = tid; p < NACRY * DF; p += 256) {
        int i = p >> 6, c = p & 63;
        float acc = b1[c];
        #pragma unroll 8
        for (int k = 0; k < DF; ++k) acc += s_src[i * DF + k] * w1[k * DF + c];
        s_h1[p] = fmaxf(acc, 0.f);
    }
    __syncthreads();
    // h2 = h1 @ w2 + b2 ; residual
    for (int p = tid; p < NACRY * DF; p += 256) {
        int i = p >> 6, c = p & 63;
        float acc = b2[c];
        #pragma unroll 8
        for (int k = 0; k < DF; ++k) acc += s_h1[i * DF + k] * w2[k * DF + c];
        s_tmp[p] = acc + s_src[p];
    }
    __syncthreads();
    if (tid < NACRY) ln_row(&s_tmp[tid * DF], &atom_tf[s_idx[tid] * DF], g2, be2);
}

// ---------------------------------------------------------------------------
// WMMA fragment conventions (16x16x32 f16, wave32):
// A (16x32 row-major): lane l holds row m=l&15; half h -> k = (l>>4)*8 + (h&7) + (h>>3)*16
// B (32x16):           lane l holds col n=l&15; half h -> k = (l>>4)*8 + (h&7) + (h>>3)*16
// C/D (16x16 f32):     lane l = col n + 16*(m>>3); vgpr v = m&7
//
// LDS weight tile is stored PRE-PACKED in fragment order:
//   frag = kt*8+nt; per (frag,lane) the 16 halves are contiguous (32B)
//   -> each B-fragment load is one aligned 32-byte LDS read (2x ds_load_b128).
// Packed k mapping for (frag,lane,h): k = kt*32 + (lane>>4)*8 + (h&7) + (h>>3)*16, c = nt*16 + (lane&15)

// cooperative loader: fc_w[kofs + k][c] (k in [0,64), zero-padded past klim) -> packed LDS
__device__ inline void load_w_packed(unsigned* s_wp32, const float* __restrict__ fc_w,
                                     int kofs, int klim, int tid) {
    // 4096 packed pairs total, 16 per thread
    for (int i = tid; i < (DF * C2) / 2; i += 256) {
        int hp   = i & 7;            // half-pair index (k pair)
        int lane = (i >> 3) & 31;
        int frag = i >> 8;           // 0..15
        int kt = frag >> 3, nt = frag & 7;
        int k0 = kt * 32 + ((lane >> 4) << 3) + 2 * (hp & 3) + ((hp >> 2) << 4);
        int c  = nt * 16 + (lane & 15);
        union { _Float16 h[2]; unsigned u; } pk;
        pk.h[0] = (k0     < klim) ? (_Float16)fc_w[(kofs + k0    ) * C2 + c] : (_Float16)0.f;
        pk.h[1] = (k0 + 1 < klim) ? (_Float16)fc_w[(kofs + k0 + 1) * C2 + c] : (_Float16)0.f;
        s_wp32[(frag * 32 + lane) * 8 + hp] = pk.u;
    }
}

__device__ inline v16h load_b_frag(const unsigned* s_wp32, int frag, int lane) {
    return *(const v16h*)(s_wp32 + (frag * 32 + lane) * 8);
}

// ---------------------------------------------------------------------------
// K2: OUT[r, 0:128] = A[r, 0:64] @ fc_w[kofs:kofs+64, 0:128] (+ bias)
__global__ void __launch_bounds__(256) k_gemm64(
    const float* __restrict__ A, const float* __restrict__ fc_w,
    const float* __restrict__ bias, int kofs, int use_bias,
    float* __restrict__ OUT, int rows)
{
    __shared__ unsigned s_wp32[(DF * C2) / 2];
    const int tid = threadIdx.x;
    load_w_packed(s_wp32, fc_w, kofs, DF, tid);
    __syncthreads();

    const int wave = tid >> 5, lane = tid & 31;
    const int r0 = (blockIdx.x * 8 + wave) * 16;
    if (r0 >= rows) return;

    const int mrow = r0 + (lane & 15);
    const int kbA  = (lane >> 4) * 8;
    const int cb   = lane & 15;

    v8f acc[8] = {};
    const float* ap = A + (long)mrow * DF;
    #pragma unroll
    for (int kt = 0; kt < 2; ++kt) {
        const int kbase = kt * 32;
        // two aligned 8-float runs: k = kbase+kbA+0..7 and +16..23
        const float4* p0 = (const float4*)(ap + kbase + kbA);
        const float4* p1 = (const float4*)(ap + kbase + kbA + 16);
        float4 x0 = p0[0], x1 = p0[1], y0 = p1[0], y1 = p1[1];
        v16h af;
        af[0]=(_Float16)x0.x; af[1]=(_Float16)x0.y; af[2]=(_Float16)x0.z; af[3]=(_Float16)x0.w;
        af[4]=(_Float16)x1.x; af[5]=(_Float16)x1.y; af[6]=(_Float16)x1.z; af[7]=(_Float16)x1.w;
        af[8]=(_Float16)y0.x; af[9]=(_Float16)y0.y; af[10]=(_Float16)y0.z; af[11]=(_Float16)y0.w;
        af[12]=(_Float16)y1.x; af[13]=(_Float16)y1.y; af[14]=(_Float16)y1.z; af[15]=(_Float16)y1.w;
        #pragma unroll
        for (int nt = 0; nt < 8; ++nt) {
            v16h bf = load_b_frag(s_wp32, kt * 8 + nt, lane);
            acc[nt] = __builtin_amdgcn_wmma_f32_16x16x32_f16(
                false, af, false, bf, (short)0, acc[nt], false, false);
        }
    }
    #pragma unroll
    for (int nt = 0; nt < 8; ++nt) {
        const int c = nt * 16 + cb;
        const float bv = use_bias ? bias[c] : 0.f;
        #pragma unroll
        for (int v = 0; v < 8; ++v) {
            const int row = r0 + v + ((lane >> 4) << 3);
            OUT[(long)row * C2 + c] = acc[nt][v] + bv;
        }
    }
}

// ---------------------------------------------------------------------------
// K3: gated[(n,m), c] = PA[n,c] + PB[idx[n,m],c] + nbr_fea[n,m,:] @ fc_w[128:169, c]
// K padded 41 -> 64 with zeros. Also accumulate BN1 sum/sumsq.
__global__ void __launch_bounds__(256) k_main(
    const float* __restrict__ nbr_fea, const float* __restrict__ fc_w,
    const int*   __restrict__ nbr_idx,
    const float* __restrict__ PA, const float* __restrict__ PB,
    _Float16* __restrict__ gated,
    float* __restrict__ gsum, float* __restrict__ gsq)
{
    __shared__ unsigned s_wp32[(DF * C2) / 2];
    __shared__ float s_sum[C2], s_sq[C2];
    const int tid = threadIdx.x;
    load_w_packed(s_wp32, fc_w, 2 * DF, NBRF, tid);
    if (tid < C2) { s_sum[tid] = 0.f; s_sq[tid] = 0.f; }
    __syncthreads();

    const int wave = tid >> 5, lane = tid & 31;
    const int r0 = (blockIdx.x * 8 + wave) * 16;   // grid sized exactly: 45000 tiles
    const int mrow = r0 + (lane & 15);
    const int kbA  = (lane >> 4) * 8;
    const int cb   = lane & 15;

    v8f acc[8] = {};
    const float* ap = nbr_fea + (long)mrow * NBRF;   // 41-float rows (unaligned: scalar loads)
    #pragma unroll
    for (int kt = 0; kt < 2; ++kt) {
        const int kbase = kt * 32;
        v16h af;
        #pragma unroll
        for (int h = 0; h < 16; ++h) {
            int kk = kbase + kbA + (h & 7) + ((h >> 3) << 4);
            af[h] = (kk < NBRF) ? (_Float16)ap[kk] : (_Float16)0.f;
        }
        #pragma unroll
        for (int nt = 0; nt < 8; ++nt) {
            v16h bf = load_b_frag(s_wp32, kt * 8 + nt, lane);
            acc[nt] = __builtin_amdgcn_wmma_f32_16x16x32_f16(
                false, af, false, bf, (short)0, acc[nt], false, false);
        }
    }
    #pragma unroll
    for (int nt = 0; nt < 8; ++nt) {
        const int c = nt * 16 + cb;
        float ssum = 0.f, ssq = 0.f;
        #pragma unroll
        for (int v = 0; v < 8; ++v) {
            const int row = r0 + v + ((lane >> 4) << 3);
            const int n = row / MNBR;
            float gv = acc[nt][v] + PA[(long)n * C2 + c]
                                  + PB[(long)nbr_idx[row] * C2 + c];
            gated[(long)row * C2 + c] = (_Float16)gv;
            ssum += gv; ssq += gv * gv;
        }
        // lanes l and l+16 hold the same channel c: combine, then one atomic
        ssum += __shfl_xor(ssum, 16, 32);
        ssq  += __shfl_xor(ssq, 16, 32);
        if (lane < 16) {
            atomicAdd(&s_sum[c], ssum);
            atomicAdd(&s_sq[c], ssq);
        }
    }
    __syncthreads();
    if (tid < C2) { atomicAdd(&gsum[tid], s_sum[tid]); atomicAdd(&gsq[tid], s_sq[tid]); }
}

// ---------------------------------------------------------------------------
// K4: finalize BN1 stats (128 channels over 720000 rows)
__global__ void k_fin1(float* __restrict__ stats) {
    int t = threadIdx.x;            // 128
    float mu  = stats[t] * (1.f / (float)NMROWS);
    float var = stats[128 + t] * (1.f / (float)NMROWS) - mu * mu;
    stats[256 + t] = mu;
    stats[384 + t] = rsqrtf(var + EPSV);
}

// ---------------------------------------------------------------------------
// K5: normalize gated, gate = sigmoid(filter)*softplus(core), sum over M,
//     accumulate BN2 stats. One thread per (atom, channel).
__global__ void __launch_bounds__(256) k_gate(
    const _Float16* __restrict__ gated, const float* __restrict__ stats,
    const float* __restrict__ bn1_g, const float* __restrict__ bn1_b,
    float* __restrict__ ns, float* __restrict__ gsum2, float* __restrict__ gsq2)
{
    __shared__ float s_sum[DF], s_sq[DF];
    const int tid = threadIdx.x;
    if (tid < DF) { s_sum[tid] = 0.f; s_sq[tid] = 0.f; }
    __syncthreads();

    const int g = blockIdx.x * 256 + tid;   // < 60000*64 exactly
    const int n = g >> 6, d = g & 63;
    const float muF = stats[256 + d],      rsF = stats[384 + d];
    const float muC = stats[256 + DF + d], rsC = stats[384 + DF + d];
    const float gF = bn1_g[d],      bF = bn1_b[d];
    const float gC = bn1_g[DF + d], bC = bn1_b[DF + d];

    float acc = 0.f;
    const _Float16* base = gated + (long)n * MNBR * C2;
    #pragma unroll
    for (int m = 0; m < MNBR; ++m) {
        float xf = (float)base[m * C2 + d];
        float xc = (float)base[m * C2 + DF + d];
        float f = (xf - muF) * rsF * gF + bF;
        float c = (xc - muC) * rsC * gC + bC;
        float sig = 1.f / (1.f + __expf(-f));
        float sp  = (c > 20.f) ? c : log1pf(__expf(c));
        acc += sig * sp;
    }
    ns[g] = acc;
    atomicAdd(&s_sum[d], acc);
    atomicAdd(&s_sq[d], acc * acc);
    __syncthreads();
    if (tid < DF) { atomicAdd(&gsum2[tid], s_sum[tid]); atomicAdd(&gsq2[tid], s_sq[tid]); }
}

// ---------------------------------------------------------------------------
// K6: finalize BN2 stats (64 channels over 60000 rows)
__global__ void k_fin2(float* __restrict__ stats) {
    int t = threadIdx.x;            // 64
    float mu  = stats[512 + t] * (1.f / (float)NATOMS);
    float var = stats[576 + t] * (1.f / (float)NATOMS) - mu * mu;
    stats[640 + t] = mu;
    stats[704 + t] = rsqrtf(var + EPSV);
}

// ---------------------------------------------------------------------------
// K7: out = softplus(atom_tf + BN2(nbr_sumed))
__global__ void __launch_bounds__(256) k_out(
    const float* __restrict__ atom_tf, const float* __restrict__ ns,
    const float* __restrict__ stats,
    const float* __restrict__ bn2_g, const float* __restrict__ bn2_b,
    float* __restrict__ out)
{
    const int g = blockIdx.x * 256 + threadIdx.x;   // < 60000*64 exactly
    const int d = g & 63;
    float x = atom_tf[g] + (ns[g] - stats[640 + d]) * stats[704 + d] * bn2_g[d] + bn2_b[d];
    out[g] = (x > 20.f) ? x : log1pf(__expf(x));
}

// ---------------------------------------------------------------------------
extern "C" void kernel_launch(void* const* d_in, const int* in_sizes, int n_in,
                              void* d_out, int out_size, void* d_ws, size_t ws_size,
                              hipStream_t stream) {
    (void)in_sizes; (void)n_in; (void)out_size; (void)ws_size;
    const float* atom_in = (const float*)d_in[0];
    const float* nbr_fea = (const float*)d_in[1];
    const float* adj     = (const float*)d_in[2];
    const float* w1      = (const float*)d_in[3];
    const float* b1      = (const float*)d_in[4];
    const float* w2      = (const float*)d_in[5];
    const float* b2      = (const float*)d_in[6];
    const float* tln1_g  = (const float*)d_in[7];
    const float* tln1_b  = (const float*)d_in[8];
    const float* tln2_g  = (const float*)d_in[9];
    const float* tln2_b  = (const float*)d_in[10];
    const float* fc_w    = (const float*)d_in[11];
    const float* fc_b    = (const float*)d_in[12];
    const float* bn1_g   = (const float*)d_in[13];
    const float* bn1_b   = (const float*)d_in[14];
    const float* bn2_g   = (const float*)d_in[15];
    const float* bn2_b   = (const float*)d_in[16];
    const int*   nbr_idx = (const int*)d_in[17];
    const int*   cry_idx = (const int*)d_in[18];

    float* ws      = (float*)d_ws;
    float* atom_tf = ws + WS_ATOMTF;
    float* PA      = ws + WS_PA;
    float* PB      = ws + WS_PB;
    float* ns      = ws + WS_NS;
    float* stats   = ws + WS_STATS;
    _Float16* gated = (_Float16*)(ws + WS_GATED);
    float* out = (float*)d_out;

    k_zero_stats<<<1, 768, 0, stream>>>(stats);
    k_transformer<<<BCRY, 256, 0, stream>>>(atom_in, adj, w1, b1, w2, b2,
                                            tln1_g, tln1_b, tln2_g, tln2_b,
                                            cry_idx, atom_tf);
    // PA = atom_tf @ fc_w[0:64] + fc_b ; PB = atom_in @ fc_w[64:128]
    k_gemm64<<<469, 256, 0, stream>>>(atom_tf, fc_w, fc_b, 0, 1, PA, NATOMS);
    k_gemm64<<<469, 256, 0, stream>>>(atom_in, fc_w, fc_b, DF, 0, PB, NATOMS);
    // gated + BN1 partial stats   (720000/16 = 45000 tiles / 8 waves = 5625 blocks)
    k_main<<<5625, 256, 0, stream>>>(nbr_fea, fc_w, nbr_idx, PA, PB,
                                     gated, stats + 0, stats + 128);
    k_fin1<<<1, 128, 0, stream>>>(stats);
    k_gate<<<15000, 256, 0, stream>>>(gated, stats, bn1_g, bn1_b,
                                      ns, stats + 512, stats + 576);
    k_fin2<<<1, 64, 0, stream>>>(stats);
    k_out<<<15000, 256, 0, stream>>>(atom_tf, ns, stats, bn2_g, bn2_b, out);
}
